// SynthesizedPrimitive_85504208929289
// MI455X (gfx1250) — matrix operand order repaired and checked by reading, not compiled, not run on hardware
//
#include <hip/hip_runtime.h>
#include <hip/hip_bf16.h>

// Problem constants (match reference)
#define BB 2
#define NN 4096
#define DD 1024
#define KK 256
#define SCALE 0.03125f  // 1/sqrt(1024)

typedef __bf16 bf16_t;
typedef bf16_t v16bf __attribute__((ext_vector_type(16)));
typedef bf16_t v8bf  __attribute__((ext_vector_type(8)));
typedef float  v8f   __attribute__((ext_vector_type(8)));

__device__ __forceinline__ v8f wmma_bf16(v16bf a, v16bf b, v8f c) {
  return __builtin_amdgcn_wmma_f32_16x16x32_bf16(
      /*neg_a=*/false, a, /*neg_b=*/false, b,
      /*c_mod=*/(short)0, c, /*reuse_a=*/false, /*reuse_b=*/false);
}

// ---- fragment loaders ------------------------------------------------------
// A (16x32 bf16, row-major source): lane m<16 holds K {0..7,16..23}, lane m+16
// holds K {8..15,24..31} of row m.
__device__ __forceinline__ v16bf load_a_bf16(const bf16_t* __restrict__ base,
                                             int ld, int row, int k0, int lane) {
  const bf16_t* r = base + (size_t)row * ld + k0 + ((lane < 16) ? 0 : 8);
  v8bf c0 = *(const v8bf*)(r);
  v8bf c1 = *(const v8bf*)(r + 16);
  v16bf out;
#pragma unroll
  for (int i = 0; i < 8; ++i) { out[i] = c0[i]; out[i + 8] = c1[i]; }
  return out;
}

// A fragment built from f32 source with on-the-fly bf16 convert.
__device__ __forceinline__ v16bf load_a_f32(const float* __restrict__ base,
                                            int ld, int row, int k0, int lane) {
  const float* r = base + (size_t)row * ld + k0 + ((lane < 16) ? 0 : 8);
  v16bf out;
#pragma unroll
  for (int i = 0; i < 8; ++i) {
    out[i]     = (bf16_t)r[i];
    out[i + 8] = (bf16_t)r[i + 16];
  }
  return out;
}

// B (32x16 bf16): column n of B = row `srow` of a row-major [*, ld] source.
// lane n<16 holds K {0..7,16..23}, lane n+16 holds K {8..15,24..31}.
__device__ __forceinline__ v16bf load_b_bf16(const bf16_t* __restrict__ base,
                                             int ld, int srow, int k0, int lane) {
  const bf16_t* r = base + (size_t)srow * ld + k0 + ((lane < 16) ? 0 : 8);
  v8bf c0 = *(const v8bf*)(r);
  v8bf c1 = *(const v8bf*)(r + 16);
  v16bf out;
#pragma unroll
  for (int i = 0; i < 8; ++i) { out[i] = c0[i]; out[i + 8] = c1[i]; }
  return out;
}

// ---- kernel 1a: weight transpose + convert: [D,K] f32 -> [K,D] bf16 --------
__global__ void wt_kernel(const float* __restrict__ wl, const float* __restrict__ wr,
                          bf16_t* __restrict__ wltT, bf16_t* __restrict__ wrtT) {
  int idx = blockIdx.x * blockDim.x + threadIdx.x;     // over K*D
  if (idx >= KK * DD) return;
  int k = idx / DD, d = idx % DD;
  wltT[idx] = (bf16_t)wl[(size_t)d * KK + k];
  wrtT[idx] = (bf16_t)wr[(size_t)d * KK + k];
}

// ---- kernel 1b: LDS-tiled transpose+convert: [B,N,D] f32 -> [B,D,N] bf16 ---
// 32x32 tiles; both global read (along D) and write (along N) are coalesced.
__global__ __launch_bounds__(256)
void xt_kernel(const float* __restrict__ x, bf16_t* __restrict__ xT) {
  __shared__ bf16_t tile[32][33];   // +1 pad to spread LDS banks
  int tx = threadIdx.x;             // 0..31
  int ty = threadIdx.y;             // 0..7
  int d0 = blockIdx.x * 32;
  int n0 = blockIdx.y * 32;
  int b  = blockIdx.z;
  const float* xb = x + (size_t)b * NN * DD;
#pragma unroll
  for (int yy = 0; yy < 4; ++yy) {
    int n = n0 + ty + yy * 8;
    tile[ty + yy * 8][tx] = (bf16_t)xb[(size_t)n * DD + d0 + tx];
  }
  __syncthreads();
  bf16_t* xTb = xT + (size_t)b * DD * NN;
#pragma unroll
  for (int yy = 0; yy < 4; ++yy) {
    int d = d0 + ty + yy * 8;
    xTb[(size_t)d * NN + n0 + tx] = tile[tx][ty + yy * 8];
  }
}

// ---- kernel 2: projections L = x@Wl, R = x@Wr (bf16 out) -------------------
__global__ __launch_bounds__(32)
void proj_kernel(const float* __restrict__ x,
                 const bf16_t* __restrict__ wltT, const bf16_t* __restrict__ wrtT,
                 bf16_t* __restrict__ Lbf, bf16_t* __restrict__ Rbf) {
  int lane = threadIdx.x;
  int k0 = blockIdx.x * 16;
  int n0 = blockIdx.y * 16;
  int b  = blockIdx.z;
  const float* xb = x + (size_t)b * NN * DD;
  int arow = n0 + (lane & 15);
  int bcol = k0 + (lane & 15);
  v8f cL = {}, cR = {};
  for (int d0 = 0; d0 < DD; d0 += 32) {
    v16bf a  = load_a_f32(xb, DD, arow, d0, lane);
    v16bf bl = load_b_bf16(wltT, DD, bcol, d0, lane);
    cL = wmma_bf16(a, bl, cL);
    v16bf br = load_b_bf16(wrtT, DD, bcol, d0, lane);
    cR = wmma_bf16(a, br, cR);
  }
  bf16_t* Lb = Lbf + (size_t)b * NN * KK;
  bf16_t* Rb = Rbf + (size_t)b * NN * KK;
  int rbase = n0 + ((lane < 16) ? 0 : 8);
#pragma unroll
  for (int v = 0; v < 8; ++v) {
    Lb[(size_t)(rbase + v) * KK + bcol] = (bf16_t)cL[v];
    Rb[(size_t)(rbase + v) * KK + bcol] = (bf16_t)cR[v];
  }
}

// ---- kernel 3: per-row softmax stats (m, l); 8 waves split the N sweep -----
__global__ __launch_bounds__(256)
void stats_kernel(const bf16_t* __restrict__ Lbf, const bf16_t* __restrict__ Rbf,
                  const float* __restrict__ gb_p,
                  float* __restrict__ mbuf, float* __restrict__ lbuf) {
  __shared__ float sm[8][16];
  __shared__ float sl[8][16];
  int tid = threadIdx.x;
  int lane = tid & 31;
  int wave = tid >> 5;             // 0..7, owns columns [wave*512, +512)
  int n0 = blockIdx.x * 16;
  int b  = blockIdx.y;
  const bf16_t* Lb = Lbf + (size_t)b * NN * KK;
  const bf16_t* Rb = Rbf + (size_t)b * NN * KK;
  float gb = gb_p[0];

  int arow = n0 + (lane & 15);
  v16bf aL[8];
#pragma unroll
  for (int kk = 0; kk < 8; ++kk) aL[kk] = load_a_bf16(Lb, KK, arow, kk * 32, lane);

  float m[8], l[8];
#pragma unroll
  for (int v = 0; v < 8; ++v) { m[v] = -3.0e38f; l[v] = 0.0f; }

  int jend = wave * 512 + 512;
  for (int j0 = wave * 512; j0 < jend; j0 += 16) {
    int brow = j0 + (lane & 15);
    v8f s = {};
#pragma unroll
    for (int kk = 0; kk < 8; ++kk) {
      v16bf bb = load_b_bf16(Rb, KK, brow, kk * 32, lane);
      s = wmma_bf16(aL[kk], bb, s);
    }
#pragma unroll
    for (int v = 0; v < 8; ++v) {
      float logit = s[v] * SCALE + gb;
      float mn = fmaxf(m[v], logit);
      l[v] = l[v] * __expf(m[v] - mn) + __expf(logit - mn);
      m[v] = mn;
    }
  }
  // butterfly reduce inside each 16-lane half (rows 0-7 / 8-15 split)
#pragma unroll
  for (int off = 1; off < 16; off <<= 1) {
#pragma unroll
    for (int v = 0; v < 8; ++v) {
      float om = __shfl_xor(m[v], off, 32);
      float ol = __shfl_xor(l[v], off, 32);
      float mn = fmaxf(m[v], om);
      l[v] = l[v] * __expf(m[v] - mn) + ol * __expf(om - mn);
      m[v] = mn;
    }
  }
  if ((lane & 15) == 0) {
    int rloc = (lane < 16) ? 0 : 8;
#pragma unroll
    for (int v = 0; v < 8; ++v) { sm[wave][rloc + v] = m[v]; sl[wave][rloc + v] = l[v]; }
  }
  __syncthreads();
  // merge the 8 per-wave partials; 16 threads, one per row
  if (tid < 16) {
    float mm = -3.0e38f, ll = 0.0f;
#pragma unroll
    for (int w = 0; w < 8; ++w) {
      float om = sm[w][tid], ol = sl[w][tid];
      float mn = fmaxf(mm, om);
      ll = ll * __expf(mm - mn) + ol * __expf(om - mn);
      mm = mn;
    }
    mbuf[(size_t)b * NN + n0 + tid] = mm;
    lbuf[(size_t)b * NN + n0 + tid] = ll;
  }
}

// ---- kernel 4: O = (mask * softmax) @ x, 8 cooperative waves ---------------
#define PSTRIDE 272  // 16x256 bf16 P slab, padded: 544B row pitch, 16B aligned
__global__ __launch_bounds__(256)
void pv_kernel(const bf16_t* __restrict__ Lbf, const bf16_t* __restrict__ Rbf,
               const bf16_t* __restrict__ xT,
               const float* __restrict__ mbuf, const float* __restrict__ lbuf,
               const float* __restrict__ thr_p, const float* __restrict__ gb_p,
               float* __restrict__ out) {
  __shared__ bf16_t pbuf[16 * PSTRIDE];
  int tid = threadIdx.x;
  int lane = tid & 31;
  int wave = tid >> 5;          // 0..7, owns D chunk [wave*128, +128)
  int n0 = blockIdx.x * 16;
  int b  = blockIdx.y;
  const bf16_t* Lb  = Lbf + (size_t)b * NN * KK;
  const bf16_t* Rb  = Rbf + (size_t)b * NN * KK;
  const bf16_t* xTb = xT  + (size_t)b * DD * NN;
  float thr = thr_p[0], gb = gb_p[0];

  int arow = n0 + (lane & 15);
  v16bf aL[8];
#pragma unroll
  for (int kk = 0; kk < 8; ++kk) aL[kk] = load_a_bf16(Lb, KK, arow, kk * 32, lane);

  int rbase = n0 + ((lane < 16) ? 0 : 8);
  float mv[8], il[8];
#pragma unroll
  for (int v = 0; v < 8; ++v) {
    mv[v] = mbuf[(size_t)b * NN + rbase + v];
    il[v] = 1.0f / lbuf[(size_t)b * NN + rbase + v];
  }

  v8f o[8];
#pragma unroll
  for (int sub = 0; sub < 8; ++sub) o[sub] = v8f{};

  int d0 = wave * 128;
  for (int j0 = 0; j0 < NN; j0 += 256) {
    int jw = j0 + wave * 32;   // this wave's 32-column slab of S/P
#pragma unroll
    for (int t = 0; t < 2; ++t) {
      int brow = jw + t * 16 + (lane & 15);
      v8f s = {};
#pragma unroll
      for (int kk = 0; kk < 8; ++kk) {
        v16bf bb = load_b_bf16(Rb, KK, brow, kk * 32, lane);
        s = wmma_bf16(aL[kk], bb, s);
      }
      int colL = wave * 32 + t * 16 + (lane & 15);
      int rloc = (lane < 16) ? 0 : 8;
#pragma unroll
      for (int v = 0; v < 8; ++v) {
        float sv = s[v] * SCALE;
        float p = (sv > thr) ? (__expf(sv + gb - mv[v]) * il[v]) : 0.0f;
        pbuf[(size_t)(rloc + v) * PSTRIDE + colL] = (bf16_t)p;
      }
    }
    __syncthreads();
    // consume shared 16x256 P slab against this wave's 128-wide D chunk
#pragma unroll
    for (int kk = 0; kk < 8; ++kk) {
      v16bf pa = load_a_bf16(pbuf, PSTRIDE, (lane & 15), kk * 32, lane);
#pragma unroll
      for (int sub = 0; sub < 8; ++sub) {
        int d = d0 + sub * 16 + (lane & 15);
        v16bf bx = load_b_bf16(xTb, NN, d, j0 + kk * 32, lane);
        o[sub] = wmma_bf16(pa, bx, o[sub]);
      }
    }
    __syncthreads();
  }
  float* outb = out + (size_t)b * NN * DD;
#pragma unroll
  for (int sub = 0; sub < 8; ++sub) {
    int d = d0 + sub * 16 + (lane & 15);
#pragma unroll
    for (int v = 0; v < 8; ++v) {
      outb[(size_t)(rbase + v) * DD + d] = o[sub][v];
    }
  }
}

// ---- host ------------------------------------------------------------------
extern "C" void kernel_launch(void* const* d_in, const int* in_sizes, int n_in,
                              void* d_out, int out_size, void* d_ws, size_t ws_size,
                              hipStream_t stream) {
  const float* x   = (const float*)d_in[0];
  const float* wl  = (const float*)d_in[1];
  const float* wr  = (const float*)d_in[2];
  const float* thr = (const float*)d_in[3];
  const float* gb  = (const float*)d_in[4];
  float* out = (float*)d_out;

  char* ws = (char*)d_ws;
  size_t off = 0;
  bf16_t* wltT = (bf16_t*)(ws + off); off += (size_t)KK * DD * sizeof(bf16_t);
  bf16_t* wrtT = (bf16_t*)(ws + off); off += (size_t)KK * DD * sizeof(bf16_t);
  bf16_t* Lbf  = (bf16_t*)(ws + off); off += (size_t)BB * NN * KK * sizeof(bf16_t);
  bf16_t* Rbf  = (bf16_t*)(ws + off); off += (size_t)BB * NN * KK * sizeof(bf16_t);
  bf16_t* xT   = (bf16_t*)(ws + off); off += (size_t)BB * DD * NN * sizeof(bf16_t);
  float*  mbuf = (float*)(ws + off);  off += (size_t)BB * NN * sizeof(float);
  float*  lbuf = (float*)(ws + off);  off += (size_t)BB * NN * sizeof(float);

  // 1) converts/transposes
  wt_kernel<<<(KK * DD + 255) / 256, 256, 0, stream>>>(wl, wr, wltT, wrtT);
  xt_kernel<<<dim3(DD / 32, NN / 32, BB), dim3(32, 8), 0, stream>>>(x, xT);

  // 2) projections
  proj_kernel<<<dim3(KK / 16, NN / 16, BB), 32, 0, stream>>>(x, wltT, wrtT, Lbf, Rbf);

  // 3) softmax stats (8 waves/block, each sweeps 512 columns, LDS merge)
  stats_kernel<<<dim3(NN / 16, BB), 256, 0, stream>>>(Lbf, Rbf, gb, mbuf, lbuf);

  // 4) masked-softmax @ x
  pv_kernel<<<dim3(NN / 16, BB), 256, 0, stream>>>(Lbf, Rbf, xT, mbuf, lbuf, thr, gb, out);
}
